// MultiHeadAttention_37847251812357
// MI455X (gfx1250) — matrix operand order repaired and checked by reading, not compiled, not run on hardware
//
#include <hip/hip_runtime.h>

typedef __bf16 bf16;
typedef __attribute__((ext_vector_type(16))) __bf16 v16bf;
typedef __attribute__((ext_vector_type(8)))  __bf16 v8bf;
typedef __attribute__((ext_vector_type(8)))  float  v8f;

#define NUM_HEADS 16
#define D_MODEL   1024
#define D_HEAD    64
#define BATCH     2
#define SEQ       2048
#define LSTR      72      // LDS row stride in bf16 (144 B: 16B-aligned, conflict-free)

// ---------------------------------------------------------------------------
// Fragments for V_WMMA_F32_16X16X32_BF16 (wave32).
// A tile in LDS: row-major [m][k].  Lane (hlf,m) loads K {0..15}+16*hlf
// contiguous -> two contiguous 16-byte ds loads.
// B tile in LDS: N-major [n][k].  Lane (hlf,n) loads K {0..15}+16*hlf
// contiguous (ascending K == fragment element order) -> two 16-byte ds loads.
// C/D: VGPR i -> M = i + 8*hlf, N = lane&15.
// ---------------------------------------------------------------------------
__device__ __forceinline__ v16bf load_frag_a(const bf16* lds, int row0, int k0) {
    const int lane = threadIdx.x & 31;
    const int hlf  = lane >> 4;
    const int m    = lane & 15;
    const bf16* p  = lds + (row0 + m) * LSTR + k0 + hlf * 16;
    v8bf lo = *(const v8bf*)(p);
    v8bf hi = *(const v8bf*)(p + 8);
    return __builtin_shufflevector(lo, hi, 0,1,2,3,4,5,6,7,8,9,10,11,12,13,14,15);
}

__device__ __forceinline__ v16bf load_frag_b(const bf16* lds, int n0, int k0) {
    const int lane = threadIdx.x & 31;
    const int hlf  = lane >> 4;
    const int n    = lane & 15;
    const bf16* p  = lds + (n0 + n) * LSTR + k0 + hlf * 16;
    v8bf lo = *(const v8bf*)(p);
    v8bf hi = *(const v8bf*)(p + 8);
    return __builtin_shufflevector(lo, hi, 0,1,2,3,4,5,6,7,8,9,10,11,12,13,14,15);
}

__device__ __forceinline__ v8f wmma_bf16(v16bf a, v16bf b, v8f c) {
    return __builtin_amdgcn_wmma_f32_16x16x32_bf16(false, a, false, b, (short)0, c,
                                                   false, false);
}

__device__ __forceinline__ void zero8(v8f& v) {
#pragma unroll
    for (int i = 0; i < 8; ++i) v[i] = 0.0f;
}

__device__ __forceinline__ void cvt_store8(bf16* dst, const float4& a, const float4& b) {
    v8bf v;
    v[0] = (bf16)a.x; v[1] = (bf16)a.y; v[2] = (bf16)a.z; v[3] = (bf16)a.w;
    v[4] = (bf16)b.x; v[5] = (bf16)b.y; v[6] = (bf16)b.z; v[7] = (bf16)b.w;
    *(v8bf*)dst = v;
}

// Stage a 64x64 fp32 tile -> bf16 LDS tile (row-major, stride LSTR), 128 thr.
__device__ __forceinline__ void stage_f32_64x64(bf16* __restrict__ dst,
                                                const float* __restrict__ src,
                                                size_t sstride) {
    const int t  = threadIdx.x;
    const int r  = t >> 1;
    const int c0 = (t & 1) * 32;
    const float4* s = (const float4*)(src + (size_t)r * sstride + c0);
    float4 f[8];
#pragma unroll
    for (int j = 0; j < 8; ++j) f[j] = s[j];
    bf16* d = dst + r * LSTR + c0;
#pragma unroll
    for (int j = 0; j < 4; ++j) cvt_store8(d + 8 * j, f[2 * j], f[2 * j + 1]);
}

// Stage a ROWS x 64 bf16 tile global -> LDS (row-major, stride LSTR).
// All global loads issued before any LDS store.
template <int ROWS, int NT>
__device__ __forceinline__ void stage_bf16(bf16* __restrict__ dst,
                                           const bf16* __restrict__ src,
                                           size_t sstride) {
    constexpr int ITER = ROWS * 8 / NT;   // v8bf chunks per thread
    v8bf f[ITER];
#pragma unroll
    for (int k = 0; k < ITER; ++k) {
        const int id = threadIdx.x + k * NT;
        const int r = id >> 3, c = (id & 7) * 8;
        f[k] = *(const v8bf*)(src + (size_t)r * sstride + c);
    }
#pragma unroll
    for (int k = 0; k < ITER; ++k) {
        const int id = threadIdx.x + k * NT;
        const int r = id >> 3, c = (id & 7) * 8;
        *(v8bf*)(dst + r * LSTR + c) = f[k];
    }
}

// ---------------------------------------------------------------------------
// Kernel 1: Q/K/V projection.  grid (SEQ/64, B*H, 3), block 128 (4 waves).
// Q,K stored [B,H,S,64]; V stored TRANSPOSED [B,H,64,S] so the attention
// kernel's P.V B-operand is a direct (N-major) copy.  Q pre-scaled by 1/8.
// ---------------------------------------------------------------------------
__global__ __launch_bounds__(128) void qkv_proj_kernel(
    const float* __restrict__ xq, const float* __restrict__ xk, const float* __restrict__ xv,
    const float* __restrict__ Wq, const float* __restrict__ Wk, const float* __restrict__ Wv,
    bf16* __restrict__ Qo, bf16* __restrict__ Ko, bf16* __restrict__ Vo)
{
    __shared__ bf16 Abuf[64 * LSTR];   // x tile [row][d]
    __shared__ bf16 Bbuf[64 * LSTR];   // W tile [e][d] (natural == N-major)

    const int z  = blockIdx.z;
    const int b  = blockIdx.y / NUM_HEADS;
    const int h  = blockIdx.y % NUM_HEADS;
    const int s0 = blockIdx.x * 64;

    const float* xs = (z == 0) ? xq : (z == 1) ? xk : xv;
    const float* Ws = (z == 0) ? Wq : (z == 1) ? Wk : Wv;
    const float scale = (z == 0) ? 0.125f : 1.0f;

    const int wave = threadIdx.x >> 5;
    const int lane = threadIdx.x & 31;
    const int hlf  = lane >> 4;
    const int n    = lane & 15;

    v8f acc[4];
#pragma unroll
    for (int nt = 0; nt < 4; ++nt) zero8(acc[nt]);

    const float* xbase = xs + (size_t)(b * SEQ + s0) * D_MODEL;
    const float* wbase = Ws + (size_t)h * D_HEAD * D_MODEL;

    for (int kc = 0; kc < D_MODEL; kc += 64) {
        stage_f32_64x64(Abuf, xbase + kc, D_MODEL);
        stage_f32_64x64(Bbuf, wbase + kc, D_MODEL);
        __syncthreads();
#pragma unroll
        for (int kk = 0; kk < 64; kk += 32) {
            const v16bf a = load_frag_a(Abuf, wave * 16, kk);
#pragma unroll
            for (int nt = 0; nt < 4; ++nt) {
                const v16bf bb = load_frag_b(Bbuf, nt * 16, kk);
                acc[nt] = wmma_bf16(a, bb, acc[nt]);
            }
        }
        __syncthreads();
    }

    const size_t base = ((size_t)(b * NUM_HEADS + h)) * SEQ * D_HEAD;
    if (z == 2) {
#pragma unroll
        for (int nt = 0; nt < 4; ++nt)
#pragma unroll
            for (int i = 0; i < 8; ++i) {
                const int row = s0 + wave * 16 + i + hlf * 8;   // s
                const int e   = nt * 16 + n;
                Vo[base + (size_t)e * SEQ + row] = (bf16)acc[nt][i];
            }
    } else {
        bf16* Oo = (z == 0) ? Qo : Ko;
#pragma unroll
        for (int nt = 0; nt < 4; ++nt)
#pragma unroll
            for (int i = 0; i < 8; ++i) {
                const int row = s0 + wave * 16 + i + hlf * 8;
                Oo[base + (size_t)row * D_HEAD + nt * 16 + n] = (bf16)(acc[nt][i] * scale);
            }
    }
}

// ---------------------------------------------------------------------------
// Kernel 2: flash attention.  grid (SEQ/128, B*H), block 256 (8 waves; wave
// owns 16 q-rows of a 128-row tile).  All LDS tiles are direct copies:
//   Qs[q][dh] (A, fragments hoisted out of the key loop),
//   Ks[key][dh] (N-major B for Q.K^T),
//   Vls[dh][key] from pre-transposed V (N-major B for P.V),
//   Ps[q][key] (A; per-wave private rows -> fence, not barrier).
// ---------------------------------------------------------------------------
__global__ __launch_bounds__(256) void flash_attn_kernel(
    const bf16* __restrict__ Q, const bf16* __restrict__ K, const bf16* __restrict__ Vt,
    bf16* __restrict__ heads)
{
    __shared__ bf16 Qs [128 * LSTR];
    __shared__ bf16 Ks [ 64 * LSTR];
    __shared__ bf16 Vls[ 64 * LSTR];
    __shared__ bf16 Ps [128 * LSTR];

    const int b  = blockIdx.y >> 4;
    const int h  = blockIdx.y & 15;
    const int s0 = blockIdx.x * 128;

    const int wave = threadIdx.x >> 5;
    const int lane = threadIdx.x & 31;
    const int hlf  = lane >> 4;
    const int n    = lane & 15;

    const size_t hbase = ((size_t)(b * NUM_HEADS + h)) * SEQ * D_HEAD;

    stage_bf16<128, 256>(Qs, Q + hbase + (size_t)s0 * D_HEAD, D_HEAD);
    __syncthreads();
    // Q fragments are invariant over the whole key loop: hoist into registers.
    const v16bf aq0 = load_frag_a(Qs, wave * 16, 0);
    const v16bf aq1 = load_frag_a(Qs, wave * 16, 32);

    v8f ofr[4];
#pragma unroll
    for (int nt = 0; nt < 4; ++nt) zero8(ofr[nt]);
    float mrow[8], lrow[8];
#pragma unroll
    for (int i = 0; i < 8; ++i) { mrow[i] = -1e30f; lrow[i] = 0.0f; }

    for (int kb = 0; kb < SEQ; kb += 64) {
        stage_bf16<64, 256>(Ks,  K  + hbase + (size_t)kb * D_HEAD, D_HEAD); // [key][dh]
        stage_bf16<64, 256>(Vls, Vt + hbase + kb,                  SEQ);    // [dh][key]
        __syncthreads();

        // S = Q.K^T (pre-scaled through Q)
        v8f sfr[4];
#pragma unroll
        for (int nt = 0; nt < 4; ++nt) {
            zero8(sfr[nt]);
            sfr[nt] = wmma_bf16(aq0, load_frag_b(Ks, nt * 16, 0),  sfr[nt]);
            sfr[nt] = wmma_bf16(aq1, load_frag_b(Ks, nt * 16, 32), sfr[nt]);
        }

        // online softmax (row stats replicated across the 16 lanes of a half)
        float mnew[8], alpha[8], rs[8];
#pragma unroll
        for (int i = 0; i < 8; ++i) {
            float v = fmaxf(fmaxf(sfr[0][i], sfr[1][i]), fmaxf(sfr[2][i], sfr[3][i]));
            v = fmaxf(v, __shfl_xor(v, 1, 16));
            v = fmaxf(v, __shfl_xor(v, 2, 16));
            v = fmaxf(v, __shfl_xor(v, 4, 16));
            v = fmaxf(v, __shfl_xor(v, 8, 16));
            mnew[i]  = fmaxf(mrow[i], v);
            alpha[i] = __expf(mrow[i] - mnew[i]);
            rs[i]    = 0.0f;
        }
#pragma unroll
        for (int nt = 0; nt < 4; ++nt)
#pragma unroll
            for (int i = 0; i < 8; ++i) {
                const float p = __expf(sfr[nt][i] - mnew[i]);
                Ps[(wave * 16 + i + hlf * 8) * LSTR + nt * 16 + n] = (bf16)p;
                rs[i] += p;
            }
#pragma unroll
        for (int i = 0; i < 8; ++i) {
            rs[i] += __shfl_xor(rs[i], 1, 16);
            rs[i] += __shfl_xor(rs[i], 2, 16);
            rs[i] += __shfl_xor(rs[i], 4, 16);
            rs[i] += __shfl_xor(rs[i], 8, 16);
            lrow[i] = lrow[i] * alpha[i] + rs[i];
            mrow[i] = mnew[i];
        }
#pragma unroll
        for (int nt = 0; nt < 4; ++nt)
#pragma unroll
            for (int i = 0; i < 8; ++i) ofr[nt][i] *= alpha[i];

        // Ps rows are private to this wave and DS ops are in-order within a
        // wave: a workgroup fence (no barrier) is enough for the reload.
        __threadfence_block();

        // O += P.V
#pragma unroll
        for (int kk = 0; kk < 64; kk += 32) {
            const v16bf a = load_frag_a(Ps, wave * 16, kk);
#pragma unroll
            for (int nt = 0; nt < 4; ++nt) {
                const v16bf bb = load_frag_b(Vls, nt * 16, kk);
                ofr[nt] = wmma_bf16(a, bb, ofr[nt]);
            }
        }
        __syncthreads();   // all waves done reading Ks/Vls before restage
    }

    const size_t obase = (size_t)b * SEQ * D_MODEL + (size_t)h * D_HEAD;
#pragma unroll
    for (int nt = 0; nt < 4; ++nt)
#pragma unroll
        for (int i = 0; i < 8; ++i) {
            const int row = s0 + wave * 16 + i + hlf * 8;
            heads[obase + (size_t)row * D_MODEL + nt * 16 + n] =
                (bf16)(ofr[nt][i] / lrow[i]);
        }
}

// ---------------------------------------------------------------------------
// Kernel 3: out = heads . Wout^T.  grid ((B*S)/64, D/64), block 128.
// ---------------------------------------------------------------------------
__global__ __launch_bounds__(128) void out_proj_kernel(
    const bf16* __restrict__ heads, const float* __restrict__ Wout,
    float* __restrict__ out)
{
    __shared__ bf16 Abuf[64 * LSTR];   // heads tile [s][d]
    __shared__ bf16 Bbuf[64 * LSTR];   // Wout tile  [o][d]  (natural == N-major)

    const int s0 = blockIdx.x * 64;
    const int o0 = blockIdx.y * 64;

    const int wave = threadIdx.x >> 5;
    const int lane = threadIdx.x & 31;
    const int hlf  = lane >> 4;
    const int n    = lane & 15;

    v8f acc[4];
#pragma unroll
    for (int nt = 0; nt < 4; ++nt) zero8(acc[nt]);

    for (int kc = 0; kc < D_MODEL; kc += 64) {
        stage_bf16<64, 128>(Abuf, heads + (size_t)s0 * D_MODEL + kc, D_MODEL);
        stage_f32_64x64(Bbuf, Wout + (size_t)o0 * D_MODEL + kc, D_MODEL);
        __syncthreads();
#pragma unroll
        for (int kk = 0; kk < 64; kk += 32) {
            const v16bf a = load_frag_a(Abuf, wave * 16, kk);
#pragma unroll
            for (int nt = 0; nt < 4; ++nt) {
                const v16bf bb = load_frag_b(Bbuf, nt * 16, kk);
                acc[nt] = wmma_bf16(a, bb, acc[nt]);
            }
        }
        __syncthreads();
    }

#pragma unroll
    for (int nt = 0; nt < 4; ++nt)
#pragma unroll
        for (int i = 0; i < 8; ++i) {
            const int row = s0 + wave * 16 + i + hlf * 8;
            out[(size_t)row * D_MODEL + o0 + nt * 16 + n] = acc[nt][i];
        }
}

// ---------------------------------------------------------------------------
extern "C" void kernel_launch(void* const* d_in, const int* in_sizes, int n_in,
                              void* d_out, int out_size, void* d_ws, size_t ws_size,
                              hipStream_t stream) {
    (void)in_sizes; (void)n_in; (void)out_size; (void)ws_size;
    const float* xq   = (const float*)d_in[0];
    const float* xk   = (const float*)d_in[1];
    const float* xv   = (const float*)d_in[2];
    const float* Wq   = (const float*)d_in[3];
    const float* Wk   = (const float*)d_in[4];
    const float* Wv   = (const float*)d_in[5];
    const float* Wout = (const float*)d_in[6];

    bf16* ws = (bf16*)d_ws;
    const size_t per = (size_t)BATCH * NUM_HEADS * SEQ * D_HEAD;  // 4M elems
    bf16* Qb = ws;
    bf16* Kb = ws + per;
    bf16* Vb = ws + 2 * per;     // transposed [B,H,64,S]
    bf16* Hd = ws + 3 * per;     // heads [B,S,D]; total 32 MB

    dim3 blk(128);
    qkv_proj_kernel<<<dim3(SEQ / 64, BATCH * NUM_HEADS, 3), blk, 0, stream>>>(
        xq, xk, xv, Wq, Wk, Wv, Qb, Kb, Vb);
    flash_attn_kernel<<<dim3(SEQ / 128, BATCH * NUM_HEADS), dim3(256), 0, stream>>>(
        Qb, Kb, Vb, Hd);
    out_proj_kernel<<<dim3((BATCH * SEQ) / 64, D_MODEL / 64), blk, 0, stream>>>(
        Hd, Wout, (float*)d_out);
}
